// NodeEdgeCrossAttention_85169201480455
// MI455X (gfx1250) — matrix operand order repaired and checked by reading, not compiled, not run on hardware
//
#include <hip/hip_runtime.h>
#include <hip/hip_bf16.h>

#define DIM   128
#define HEADS 8
#define DH    16
#define SCALE 0.25f   // DH^-0.5 = 16^-0.5

typedef __attribute__((ext_vector_type(16))) __bf16 bf16x16;
typedef __attribute__((ext_vector_type(8)))  __bf16 bf16x8;
typedef __attribute__((ext_vector_type(8)))  float  v8f;

// ---------------------------------------------------------------- helpers

__device__ __forceinline__ void atomicMaxFloat(float* addr, float val) {
    // ordered-int trick; mbuf initialized to -inf (0xFF800000)
    if (val >= 0.0f) atomicMax((int*)addr, __float_as_int(val));
    else             atomicMin((unsigned int*)addr, __float_as_uint(val));
}

// -------------------------------------------------- weight fp32 -> bf16

__global__ void convert_weights_kernel(const float* __restrict__ Wq,
                                       const float* __restrict__ Wk,
                                       const float* __restrict__ Wv,
                                       const float* __restrict__ Wo,
                                       __bf16* __restrict__ out) {
    int i = blockIdx.x * blockDim.x + threadIdx.x;          // 0 .. 4*16384-1
    if (i >= 4 * DIM * DIM) return;
    int which = i >> 14;                                    // DIM*DIM == 16384
    int j     = i & (DIM * DIM - 1);
    float v;
    if      (which == 0) v = Wq[j];
    else if (which == 1) v = Wk[j];
    else if (which == 2) v = Wv[j];
    else                 v = Wo[j];
    out[i] = (__bf16)v;
}

// -------------------------------------------------- init m/s/aggr

__global__ void init_buffers_kernel(float* __restrict__ mbuf,
                                    float* __restrict__ sbuf,
                                    float* __restrict__ aggr, int n) {
    int i = blockIdx.x * blockDim.x + threadIdx.x;
    if (i < n * DIM)   aggr[i] = 0.0f;
    if (i < n * HEADS) { mbuf[i] = -__builtin_inff(); sbuf[i] = 0.0f; }
}

// -------------------------------------------------- WMMA projection
// Out[m, o] = sum_k X[m,k] * W[o,k] + bias[o]
// One wave computes a 16x128 output tile: 4 k-steps x 8 n-tiles of
// v_wmma_f32_16x16x32_bf16 with fp32 accumulation.
//
// The 32KB weight matrix is staged once per workgroup into LDS in
// *fragment order*: frag(kidx,h) is a 1KB block where lane l's 16 bf16
// (32B) are contiguous -> conflict-free ds_load_b128 pairs. All 8 B-frags
// of a k-step are preloaded into registers so the 8 WMMAs issue
// back-to-back with no per-WMMA LDS round trip.

template <bool OUT_BF16>
__global__ void proj_wmma_kernel(const float* __restrict__ X,
                                 const __bf16* __restrict__ Wb,   // [128,128] bf16, row = out channel
                                 const float* __restrict__ bias,
                                 void* __restrict__ outp, int M) {
    __shared__ __bf16 ldsW[DIM * DIM];                   // 32 KB of the 320 KB WGP LDS

    // ---- cooperative fill, frag-order swizzle ----
    // chunk c (16B = 8 bf16): j0=(c&1)*8, lane l=(c>>1)&31, h=(c>>6)&7, kidx=c>>9
    // source element (B[k,n] = W[n,k]): row = h*16 + (l&15), col = kidx*32 + (l>>4)*16 + j0
    for (int c = threadIdx.x; c < 2048; c += blockDim.x) {
        const int j0   = (c & 1) * 8;
        const int l    = (c >> 1) & 31;
        const int h    = (c >> 6) & 7;
        const int kidx = c >> 9;
        const __bf16* src = Wb + (size_t)(h * 16 + (l & 15)) * DIM
                               + kidx * 32 + (l >> 4) * 16 + j0;
        bf16x8 v = *(const bf16x8*)src;
        *(bf16x8*)&ldsW[(size_t)((kidx * 8 + h) * 32 + l) * 16 + j0] = v;
    }
    __syncthreads();

    const int wave = threadIdx.x >> 5;
    const int lane = threadIdx.x & 31;
    const int tile = blockIdx.x * (blockDim.x >> 5) + wave;
    const int numTiles = (M + 15) >> 4;
    if (tile >= numTiles) return;                 // wave-uniform: EXEC all-ones inside

    const int r0   = tile << 4;
    const int m    = lane & 15;
    const int half = lane >> 4;
    int rowA = r0 + m; if (rowA >= M) rowA = M - 1;
    const float* __restrict__ xr = X + (size_t)rowA * DIM;

    v8f c[HEADS];
#pragma unroll
    for (int h = 0; h < HEADS; ++h)
#pragma unroll
        for (int r = 0; r < 8; ++r) c[h][r] = 0.0f;

#pragma unroll
    for (int k0 = 0; k0 < DIM; k0 += 32) {
        // A-frag (16x32 bf16): lane<16 holds K {0..15}, lane>=16 holds {16..31}
        float4 f0 = *(const float4*)(xr + k0 + half * 16 + 0);
        float4 f1 = *(const float4*)(xr + k0 + half * 16 + 4);
        float4 f2 = *(const float4*)(xr + k0 + half * 16 + 8);
        float4 f3 = *(const float4*)(xr + k0 + half * 16 + 12);
        bf16x16 a;
        a[0]  = (__bf16)f0.x; a[1]  = (__bf16)f0.y; a[2]  = (__bf16)f0.z; a[3]  = (__bf16)f0.w;
        a[4]  = (__bf16)f1.x; a[5]  = (__bf16)f1.y; a[6]  = (__bf16)f1.z; a[7]  = (__bf16)f1.w;
        a[8]  = (__bf16)f2.x; a[9]  = (__bf16)f2.y; a[10] = (__bf16)f2.z; a[11] = (__bf16)f2.w;
        a[12] = (__bf16)f3.x; a[13] = (__bf16)f3.y; a[14] = (__bf16)f3.z; a[15] = (__bf16)f3.w;

        // preload all 8 B-frags for this k-step (batched ds_load_b128 pairs)
        bf16x16 b[HEADS];
#pragma unroll
        for (int h = 0; h < HEADS; ++h)
            b[h] = *(const bf16x16*)&ldsW[(size_t)(((k0 >> 5) * 8 + h) * 32 + lane) * 16];

        // 8 back-to-back WMMAs: distinct B and C registers, shared A
#pragma unroll
        for (int h = 0; h < HEADS; ++h)
            c[h] = __builtin_amdgcn_wmma_f32_16x16x32_bf16(
                       false, a, false, b[h], (short)0, c[h], false, false);
    }

    // C/D layout: VGPR r, lanes 0-15 -> row r, lanes 16-31 -> row 8+r; col = lane&15
#pragma unroll
    for (int h = 0; h < HEADS; ++h) {
        const int col = h * 16 + m;
        const float bv = bias[col];
#pragma unroll
        for (int r = 0; r < 8; ++r) {
            const int row = r0 + r + half * 8;
            if (row < M) {
                const float val = c[h][r] + bv;
                if (OUT_BF16) ((__bf16*)outp)[(size_t)row * DIM + col] = (__bf16)val;
                else          ((float*)outp)[(size_t)row * DIM + col]  = val;
            }
        }
    }
}

// -------------------------------------------------- per-edge scores + segment max

__global__ void score_kernel(const __bf16* __restrict__ Kp,
                             const float* __restrict__ Qp,
                             const long long* __restrict__ dst_idx,
                             float* __restrict__ score,
                             float* __restrict__ mbuf, int E) {
    int e = blockIdx.x * blockDim.x + threadIdx.x;
    if (e >= E) return;
    const long long d = dst_idx[e];
    const __bf16* __restrict__ kr = Kp + (size_t)e * DIM;
    const float*  __restrict__ qr = Qp + (size_t)d * DIM;   // L2-resident gather
    float* __restrict__ sr = score + (size_t)e * HEADS;
#pragma unroll
    for (int h = 0; h < HEADS; ++h) {
        bf16x8 k0 = *(const bf16x8*)(kr + h * 16);
        bf16x8 k1 = *(const bf16x8*)(kr + h * 16 + 8);
        float4 q0 = *(const float4*)(qr + h * 16 + 0);
        float4 q1 = *(const float4*)(qr + h * 16 + 4);
        float4 q2 = *(const float4*)(qr + h * 16 + 8);
        float4 q3 = *(const float4*)(qr + h * 16 + 12);
        float s = 0.0f;
        s += (float)k0[0] * q0.x + (float)k0[1] * q0.y + (float)k0[2] * q0.z + (float)k0[3] * q0.w;
        s += (float)k0[4] * q1.x + (float)k0[5] * q1.y + (float)k0[6] * q1.z + (float)k0[7] * q1.w;
        s += (float)k1[0] * q2.x + (float)k1[1] * q2.y + (float)k1[2] * q2.z + (float)k1[3] * q2.w;
        s += (float)k1[4] * q3.x + (float)k1[5] * q3.y + (float)k1[6] * q3.z + (float)k1[7] * q3.w;
        s *= SCALE;
        sr[h] = s;
        atomicMaxFloat(&mbuf[(size_t)d * HEADS + h], s);
    }
}

// -------------------------------------------------- exp + segment sum

__global__ void expsum_kernel(float* __restrict__ score,
                              const float* __restrict__ mbuf,
                              float* __restrict__ sbuf,
                              const long long* __restrict__ dst_idx, int E) {
    int i = blockIdx.x * blockDim.x + threadIdx.x;
    if (i >= E * HEADS) return;
    int e = i >> 3, h = i & 7;
    long long d = dst_idx[e];
    float ex = __expf(score[i] - mbuf[(size_t)d * HEADS + h]);
    score[i] = ex;
    atomicAdd(&sbuf[(size_t)d * HEADS + h], ex);
}

// -------------------------------------------------- weighted scatter-aggregate
// 4 threads per edge, 32 dims (2 heads) each.

__global__ void aggregate_kernel(const __bf16* __restrict__ Vp,
                                 const float* __restrict__ score,
                                 const float* __restrict__ sbuf,
                                 const long long* __restrict__ dst_idx,
                                 float* __restrict__ aggr, int E) {
    int t = blockIdx.x * blockDim.x + threadIdx.x;
    if (t >= E * 4) return;
    int e = t >> 2, qd = t & 3;
    long long d = dst_idx[e];
    int h0 = qd * 2;
    float w0 = score[(size_t)e * HEADS + h0]     / (sbuf[(size_t)d * HEADS + h0]     + 1e-16f);
    float w1 = score[(size_t)e * HEADS + h0 + 1] / (sbuf[(size_t)d * HEADS + h0 + 1] + 1e-16f);
    const __bf16* __restrict__ vr = Vp + (size_t)e * DIM + qd * 32;
    float* __restrict__ ar = aggr + (size_t)d * DIM + qd * 32;
#pragma unroll
    for (int j = 0; j < 16; ++j) atomicAdd(&ar[j],      (float)vr[j]      * w0);
#pragma unroll
    for (int j = 0; j < 16; ++j) atomicAdd(&ar[16 + j], (float)vr[16 + j] * w1);
}

// -------------------------------------------------- host launch

extern "C" void kernel_launch(void* const* d_in, const int* in_sizes, int n_in,
                              void* d_out, int out_size, void* d_ws, size_t ws_size,
                              hipStream_t stream) {
    const float*     q_nodes = (const float*)d_in[0];
    const float*     k_edges = (const float*)d_in[1];
    const float*     v_edges = (const float*)d_in[2];
    const long long* edge_ix = (const long long*)d_in[3];  // int64, row 0 = dst
    const float* Wq = (const float*)d_in[4];  const float* bq = (const float*)d_in[5];
    const float* Wk = (const float*)d_in[6];  const float* bk = (const float*)d_in[7];
    const float* Wv = (const float*)d_in[8];  const float* bv = (const float*)d_in[9];
    const float* Wo = (const float*)d_in[10]; const float* bo = (const float*)d_in[11];

    const int N = in_sizes[0] / DIM;
    const int E = in_sizes[1] / DIM;

    // workspace carve-up (256B aligned)
    char* ws = (char*)d_ws;
    size_t off = 0;
    auto carve = [&](size_t bytes) -> char* {
        char* p = ws + off;
        off += (bytes + 255) & ~(size_t)255;
        return p;
    };
    __bf16* Wbf   = (__bf16*)carve((size_t)4 * DIM * DIM * sizeof(__bf16));
    float*  Qp    = (float*) carve((size_t)N * DIM * sizeof(float));
    __bf16* Kp    = (__bf16*)carve((size_t)E * DIM * sizeof(__bf16));
    __bf16* Vp    = (__bf16*)carve((size_t)E * DIM * sizeof(__bf16));
    float*  score = (float*) carve((size_t)E * HEADS * sizeof(float));
    float*  mbuf  = (float*) carve((size_t)N * HEADS * sizeof(float));
    float*  sbuf  = (float*) carve((size_t)N * HEADS * sizeof(float));
    float*  aggr  = (float*) carve((size_t)N * DIM * sizeof(float));
    (void)ws_size; (void)n_in; (void)out_size;

    // 1) weights -> bf16 (L2-resident, 128KB total)
    convert_weights_kernel<<<(4 * DIM * DIM + 255) / 256, 256, 0, stream>>>(Wq, Wk, Wv, Wo, Wbf);

    // 2) init segment buffers
    init_buffers_kernel<<<((size_t)N * DIM + 255) / 256, 256, 0, stream>>>(mbuf, sbuf, aggr, N);

    // 3) WMMA projections (8 waves / block, one 16x128 tile per wave, weights in LDS)
    const int WPB = 8;
    int qTiles = (N + 15) / 16;
    int eTiles = (E + 15) / 16;
    proj_wmma_kernel<false><<<(qTiles + WPB - 1) / WPB, 32 * WPB, 0, stream>>>(
        q_nodes, Wbf + 0 * DIM * DIM, bq, (void*)Qp, N);
    proj_wmma_kernel<true><<<(eTiles + WPB - 1) / WPB, 32 * WPB, 0, stream>>>(
        k_edges, Wbf + 1 * DIM * DIM, bk, (void*)Kp, E);
    proj_wmma_kernel<true><<<(eTiles + WPB - 1) / WPB, 32 * WPB, 0, stream>>>(
        v_edges, Wbf + 2 * DIM * DIM, bv, (void*)Vp, E);

    // 4) per-edge scores + segment max
    score_kernel<<<(E + 255) / 256, 256, 0, stream>>>(Kp, Qp, edge_ix, score, mbuf, E);

    // 5) exp + segment sum
    expsum_kernel<<<((size_t)E * HEADS + 255) / 256, 256, 0, stream>>>(score, mbuf, sbuf, edge_ix, E);

    // 6) weighted scatter-aggregate
    aggregate_kernel<<<((size_t)E * 4 + 255) / 256, 256, 0, stream>>>(Vp, score, sbuf, edge_ix, aggr, E);

    // 7) output projection -> d_out (fp32)
    proj_wmma_kernel<false><<<(qTiles + WPB - 1) / WPB, 32 * WPB, 0, stream>>>(
        aggr, Wbf + 3 * DIM * DIM, bo, d_out, N);
}
